// NeuralSDEForecaster_10127532884075
// MI455X (gfx1250) — compile-verified, matrix operand order checked by reading, not run on hardware
//
#include <hip/hip_runtime.h>
#include <math.h>

typedef __attribute__((ext_vector_type(16))) _Float16 v16h;
typedef __attribute__((ext_vector_type(8)))  _Float16 v8h;
typedef __attribute__((ext_vector_type(8)))  float    v8f;
typedef __attribute__((ext_vector_type(4)))  unsigned int u32x4;
typedef __attribute__((ext_vector_type(4)))  int      i32x4;
typedef __attribute__((ext_vector_type(8)))  int      i32x8;

#define HDIM 512

// TDM path only when the device compiler exposes the builtins.
#if defined(__HIP_DEVICE_COMPILE__) && defined(__has_builtin)
#if __has_builtin(__builtin_amdgcn_tensor_load_to_lds) && \
    __has_builtin(__builtin_amdgcn_s_wait_tensorcnt)
#define TDM_PATH 1
#endif
#endif
#ifndef TDM_PATH
#define TDM_PATH 0
#endif

// ---------------------------------------------------------------------------
// Fragment loader for 16-bit A/B operands of v_wmma_f32_16x16x32_f16.
// A-matrix 16x32 layout (ISA 7.12.2): lanes 0-15 hold row m=lane with
// k in {0..7}+{16..23}; lanes 16-31 hold k in {8..15}+{24..31}.
// Works for global or LDS base pointers (LDS -> ds_load_b128).
// ---------------------------------------------------------------------------
__device__ __forceinline__ v16h load_frag16(const _Float16* __restrict__ base,
                                            int row0, int k0, int ld, int lane) {
  const int r  = row0 + (lane & 15);
  const int kk = k0 + ((lane >> 4) << 3);           // 0 or 8
  const _Float16* p = base + (size_t)r * ld + kk;
  v8h lo = *(const v8h*)(p);                        // k = kk .. kk+7
  v8h hi = *(const v8h*)(p + 16);                   // k = kk+16 .. kk+23
  v16h out;
#pragma unroll
  for (int i = 0; i < 8; ++i) { out[i] = lo[i]; out[i + 8] = hi[i]; }
  return out;
}

__device__ __forceinline__ float sigmoidf_(float x) { return 1.0f / (1.0f + __expf(-x)); }
__device__ __forceinline__ float softplusf_(float x) {
  return fmaxf(x, 0.0f) + log1pf(__expf(-fabsf(x)));
}

#if TDM_PATH
// ---------------------------------------------------------------------------
// Issue one 2D TDM tile load: tile_rows x tile_k (f16) from a row-major
// matrix with row stride `stride_elems`, into LDS at byte offset lds_off.
// D# packing per CDNA5 ISA ch.8 (group0: count/lds/global/type,
// group1: data_size, tensor dims, tile dims, dim0 stride).
// ---------------------------------------------------------------------------
__device__ __forceinline__ void tdm_load_2d(const void* gsrc, unsigned lds_off,
                                            unsigned tile_k, unsigned tile_rows,
                                            unsigned stride_elems) {
  const unsigned long long ga = (unsigned long long)gsrc;
  u32x4 g0 = {};
  g0[0] = 1u;                                              // count=1 (user D#)
  g0[1] = lds_off;                                         // lds_addr (bytes)
  g0[2] = (unsigned)ga;                                    // global_addr[31:0]
  g0[3] = (unsigned)((ga >> 32) & 0x01FFFFFFu) | (2u << 30); // addr[56:32], type=2

  i32x8 g1 = {};
  const unsigned T0 = 1u << 20, T1 = 1u << 20;             // huge tensor dims (no clip)
  g1[0] = (int)(1u << 16);                                 // data_size = 2 bytes
  g1[1] = (int)((T0 & 0xFFFFu) << 16);                     // tensor_dim0[15:0]
  g1[2] = (int)(((T0 >> 16) & 0xFFFFu) | ((T1 & 0xFFFFu) << 16));
  g1[3] = (int)(((T1 >> 16) & 0xFFFFu) | ((tile_k & 0xFFFFu) << 16)); // tile_dim0
  g1[4] = (int)(tile_rows & 0xFFFFu);                      // tile_dim1 (tile_dim2=0)
  g1[5] = (int)stride_elems;                               // tensor_dim0_stride[31:0]
  // g1[6], g1[7]: stride high bits / dim1 stride = 0 (2D tile)

  i32x4 gz = {};
#if __clang_major__ >= 23
  i32x8 z8 = {};
  __builtin_amdgcn_tensor_load_to_lds(g0, g1, gz, gz, z8, 0);
#else
  __builtin_amdgcn_tensor_load_to_lds(g0, g1, gz, gz, 0);
#endif
}
#endif  // TDM_PATH

// ---------------------------------------------------------------------------
// Utility kernels
// ---------------------------------------------------------------------------
__global__ void k_f32_to_f16(const float* __restrict__ s, _Float16* __restrict__ d, int n) {
  int i = blockIdx.x * blockDim.x + threadIdx.x;
  if (i < n) d[i] = (_Float16)s[i];
}

// xT[t][b][i] = x[b][i][t]   (x: [128,32,96] -> xT: [96,128,32])
__global__ void k_x_transpose(const float* __restrict__ x, _Float16* __restrict__ xT) {
  int idx = blockIdx.x * blockDim.x + threadIdx.x;
  if (idx >= 96 * 128 * 32) return;
  int i = idx & 31;
  int b = (idx >> 5) & 127;
  int t = idx >> 12;
  xT[idx] = (_Float16)x[((size_t)b * 32 + i) * 96 + t];
}

__global__ void k_zero_f16(_Float16* __restrict__ d, int n) {
  int i = blockIdx.x * blockDim.x + threadIdx.x;
  if (i < n) d[i] = (_Float16)0.0f;
}

// dst[m][:] = src[m & 127][:]   over m in [0, 2048)
__global__ void k_broadcast(const _Float16* __restrict__ src, _Float16* __restrict__ dst) {
  int i = blockIdx.x * blockDim.x + threadIdx.x;   // i = m*512 + j
  int m = i >> 9;
  int j = i & 511;
  dst[i] = src[((size_t)(m & 127) << 9) + j];
}

// ---------------------------------------------------------------------------
// Generic fused GRU step (direct-global operand fetch).  Used for the
// encoder (M=128, with input-projection GEMM over K=32).
// ---------------------------------------------------------------------------
template <bool HAS_X, int MF, int NF, int WAVES_M, int WAVES_N>
__global__ __launch_bounds__(32 * WAVES_M * WAVES_N)
void k_gru_step(const _Float16* __restrict__ h, _Float16* __restrict__ hout,
                const _Float16* __restrict__ Whh,     // [3H, H] row-major f16
                const float* __restrict__ bih, const float* __restrict__ bhh,
                const _Float16* __restrict__ xt,      // [M, 32] f16 (or null)
                const _Float16* __restrict__ Wih) {   // [3H, 32] f16 (or null)
  const int lane = threadIdx.x & 31;
  const int w    = threadIdx.x >> 5;
  const int wm   = w % WAVES_M;
  const int wn   = w / WAVES_M;
  const int row0 = (blockIdx.y * WAVES_M + wm) * (MF * 16);
  const int n0   = (blockIdx.x * WAVES_N + wn) * (NF * 16);

  v8f acc[3][MF][NF] = {};

  for (int k0 = 0; k0 < HDIM; k0 += 32) {
    if (k0 + 32 < HDIM)
      __builtin_prefetch(h + (size_t)(row0 + (lane & 15)) * HDIM + k0 + 32, 0, 0);
    v16h a[MF];
#pragma unroll
    for (int mf = 0; mf < MF; ++mf) a[mf] = load_frag16(h, row0 + mf * 16, k0, HDIM, lane);
#pragma unroll
    for (int g = 0; g < 3; ++g) {
#pragma unroll
      for (int nf = 0; nf < NF; ++nf) {
        v16h bf = load_frag16(Whh, g * HDIM + n0 + nf * 16, k0, HDIM, lane);
#pragma unroll
        for (int mf = 0; mf < MF; ++mf)
          acc[g][mf][nf] = __builtin_amdgcn_wmma_f32_16x16x32_f16(
              false, a[mf], false, bf, (short)0, acc[g][mf][nf], false, false);
      }
    }
  }

  v8f xacc[3][MF][NF] = {};
  if (HAS_X) {
    v16h ax[MF];
#pragma unroll
    for (int mf = 0; mf < MF; ++mf) ax[mf] = load_frag16(xt, row0 + mf * 16, 0, 32, lane);
#pragma unroll
    for (int g = 0; g < 3; ++g) {
#pragma unroll
      for (int nf = 0; nf < NF; ++nf) {
        v16h bf = load_frag16(Wih, g * HDIM + n0 + nf * 16, 0, 32, lane);
#pragma unroll
        for (int mf = 0; mf < MF; ++mf)
          xacc[g][mf][nf] = __builtin_amdgcn_wmma_f32_16x16x32_f16(
              false, ax[mf], false, bf, (short)0, xacc[g][mf][nf], false, false);
      }
    }
  }

#pragma unroll
  for (int mf = 0; mf < MF; ++mf) {
#pragma unroll
    for (int nf = 0; nf < NF; ++nf) {
      const int n = n0 + nf * 16 + (lane & 15);
      const float bir = bih[n], biz = bih[n + HDIM], binn = bih[n + 2 * HDIM];
      const float bhr = bhh[n], bhz = bhh[n + HDIM], bhn  = bhh[n + 2 * HDIM];
#pragma unroll
      for (int e = 0; e < 8; ++e) {
        const int m = row0 + mf * 16 + e + ((lane >> 4) << 3);
        float ir = bir, iz = biz, inn = binn;
        if (HAS_X) {
          ir  += xacc[0][mf][nf][e];
          iz  += xacc[1][mf][nf][e];
          inn += xacc[2][mf][nf][e];
        }
        const float r  = sigmoidf_(ir + acc[0][mf][nf][e] + bhr);
        const float z  = sigmoidf_(iz + acc[1][mf][nf][e] + bhz);
        const float nn = tanhf(inn + r * (acc[2][mf][nf][e] + bhn));
        const float ho = (float)h[(size_t)m * HDIM + n];
        hout[(size_t)m * HDIM + n] = (_Float16)((1.0f - z) * nn + z * ho);
      }
    }
  }
}

// ---------------------------------------------------------------------------
// Decoder GRU step with TDM double-buffered LDS staging.
// WG = 8 waves = 4(M) x 2(N); tile 128M x 64N x 3 gates; K chunks of 32.
// Wave 0 DMAs the next chunk (A slab + 3 gate weight slabs) into LDS with
// tensor_load_to_lds while all waves run WMMA on the current chunk.
// ---------------------------------------------------------------------------
__global__ __launch_bounds__(256)
void k_gru_step_tdm(const _Float16* __restrict__ h, _Float16* __restrict__ hout,
                    const _Float16* __restrict__ Whh,
                    const float* __restrict__ bih, const float* __restrict__ bhh) {
  __shared__ _Float16 shA[2][128][32];       // 16 KB
  __shared__ _Float16 shB[2][3][64][32];     // 24 KB

  const int lane  = threadIdx.x & 31;
  const int w     = threadIdx.x >> 5;
  const int wm    = w & 3;                   // 4 waves along M
  const int wn    = w >> 2;                  // 2 waves along N
  const int rowWG = blockIdx.y * 128;
  const int nWG   = blockIdx.x * 64;
  const int mrel0 = wm * 32;
  const int nrel0 = wn * 32;

  v8f acc[3][2][2] = {};

#if TDM_PATH
  // Chunk staging via Tensor Data Mover (4 descriptors per chunk).
  auto issue = [&](int buf, int k0) {
    tdm_load_2d(h + (size_t)rowWG * HDIM + k0,
                (unsigned)(uintptr_t)&shA[buf][0][0], 32, 128, HDIM);
#pragma unroll
    for (int g = 0; g < 3; ++g)
      tdm_load_2d(Whh + (size_t)(g * HDIM + nWG) * HDIM + k0,
                  (unsigned)(uintptr_t)&shB[buf][g][0][0], 32, 64, HDIM);
  };
  if (w == 0) issue(0, 0);
#else
  // Fallback staging: cooperative vector copies (global -> LDS).
  auto issue = [&](int buf, int k0) {
    const int tid = threadIdx.x;
#pragma unroll
    for (int j = 0; j < 2; ++j) {            // A: 512 x 16B chunks
      const int c   = tid * 2 + j;
      const int row = c >> 2, kp = (c & 3) * 8;
      *(v8h*)&shA[buf][row][kp] = *(const v8h*)(h + (size_t)(rowWG + row) * HDIM + k0 + kp);
    }
#pragma unroll
    for (int j = 0; j < 3; ++j) {            // B: 768 x 16B chunks
      const int c   = tid * 3 + j;
      const int g   = c >> 8;
      const int row = (c & 255) >> 2, kp = (c & 3) * 8;
      *(v8h*)&shB[buf][g][row][kp] =
          *(const v8h*)(Whh + (size_t)(g * HDIM + nWG + row) * HDIM + k0 + kp);
    }
  };
  issue(0, 0);
#endif

  for (int kc = 0; kc < 16; ++kc) {
    const int buf = kc & 1;
#if TDM_PATH
    if (w == 0) {
      if (kc + 1 < 16) {
        issue(buf ^ 1, (kc + 1) * 32);
        __builtin_amdgcn_s_wait_tensorcnt((short)4);   // prev chunk complete
      } else {
        __builtin_amdgcn_s_wait_tensorcnt((short)0);
      }
    }
#else
    if (kc + 1 < 16) issue(buf ^ 1, (kc + 1) * 32);
#endif
    __syncthreads();                                   // chunk kc visible

    v16h a[2];
#pragma unroll
    for (int mf = 0; mf < 2; ++mf)
      a[mf] = load_frag16(&shA[buf][0][0], mrel0 + mf * 16, 0, 32, lane);
#pragma unroll
    for (int g = 0; g < 3; ++g) {
#pragma unroll
      for (int nf = 0; nf < 2; ++nf) {
        v16h bf = load_frag16(&shB[buf][g][0][0], nrel0 + nf * 16, 0, 32, lane);
#pragma unroll
        for (int mf = 0; mf < 2; ++mf)
          acc[g][mf][nf] = __builtin_amdgcn_wmma_f32_16x16x32_f16(
              false, a[mf], false, bf, (short)0, acc[g][mf][nf], false, false);
      }
    }
    __syncthreads();                                   // done reading buf
  }

#pragma unroll
  for (int mf = 0; mf < 2; ++mf) {
#pragma unroll
    for (int nf = 0; nf < 2; ++nf) {
      const int n = nWG + nrel0 + nf * 16 + (lane & 15);
      const float bir = bih[n], biz = bih[n + HDIM], binn = bih[n + 2 * HDIM];
      const float bhr = bhh[n], bhz = bhh[n + HDIM], bhn  = bhh[n + 2 * HDIM];
#pragma unroll
      for (int e = 0; e < 8; ++e) {
        const int m = rowWG + mrel0 + mf * 16 + e + ((lane >> 4) << 3);
        const float r  = sigmoidf_(bir + acc[0][mf][nf][e] + bhr);
        const float z  = sigmoidf_(biz + acc[1][mf][nf][e] + bhz);
        const float nn = tanhf(binn + r * (acc[2][mf][nf][e] + bhn));
        const float ho = (float)h[(size_t)m * HDIM + n];
        hout[(size_t)m * HDIM + n] = (_Float16)((1.0f - z) * nn + z * ho);
      }
    }
  }
}

// ---------------------------------------------------------------------------
// SDE stage 1:  u = tanh(h @ W1^T + b1)
// ---------------------------------------------------------------------------
template <int MF, int NF, int WAVES_M, int WAVES_N>
__global__ __launch_bounds__(32 * WAVES_M * WAVES_N)
void k_sde_u(const _Float16* __restrict__ h, _Float16* __restrict__ u,
             const _Float16* __restrict__ W1, const float* __restrict__ b1) {
  const int lane = threadIdx.x & 31;
  const int w    = threadIdx.x >> 5;
  const int wm   = w % WAVES_M;
  const int wn   = w / WAVES_M;
  const int row0 = (blockIdx.y * WAVES_M + wm) * (MF * 16);
  const int n0   = (blockIdx.x * WAVES_N + wn) * (NF * 16);

  v8f acc[MF][NF] = {};
  for (int k0 = 0; k0 < HDIM; k0 += 32) {
    v16h a[MF];
#pragma unroll
    for (int mf = 0; mf < MF; ++mf) a[mf] = load_frag16(h, row0 + mf * 16, k0, HDIM, lane);
#pragma unroll
    for (int nf = 0; nf < NF; ++nf) {
      v16h bf = load_frag16(W1, n0 + nf * 16, k0, HDIM, lane);
#pragma unroll
      for (int mf = 0; mf < MF; ++mf)
        acc[mf][nf] = __builtin_amdgcn_wmma_f32_16x16x32_f16(
            false, a[mf], false, bf, (short)0, acc[mf][nf], false, false);
    }
  }
#pragma unroll
  for (int mf = 0; mf < MF; ++mf) {
#pragma unroll
    for (int nf = 0; nf < NF; ++nf) {
      const int n = n0 + nf * 16 + (lane & 15);
      const float bb = b1[n];
#pragma unroll
      for (int e = 0; e < 8; ++e) {
        const int m = row0 + mf * 16 + e + ((lane >> 4) << 3);
        u[(size_t)m * HDIM + n] = (_Float16)tanhf(acc[mf][nf][e] + bb);
      }
    }
  }
}

// ---------------------------------------------------------------------------
// SDE stage 2:  h_out = h + dt*(u @ W2^T + b2) + softplus(h @ Wg^T + bg) * dW
// ---------------------------------------------------------------------------
template <int MF, int NF, int WAVES_M, int WAVES_N>
__global__ __launch_bounds__(32 * WAVES_M * WAVES_N)
void k_sde_h(const _Float16* __restrict__ h, const _Float16* __restrict__ u,
             _Float16* __restrict__ hout,
             const _Float16* __restrict__ W2, const float* __restrict__ b2,
             const _Float16* __restrict__ Wg, const float* __restrict__ bg,
             const float* __restrict__ noise, int t, float dtv, float sqdt) {
  const int lane = threadIdx.x & 31;
  const int w    = threadIdx.x >> 5;
  const int wm   = w % WAVES_M;
  const int wn   = w / WAVES_M;
  const int row0 = (blockIdx.y * WAVES_M + wm) * (MF * 16);
  const int n0   = (blockIdx.x * WAVES_N + wn) * (NF * 16);

  v8f accF[MF][NF] = {};
  v8f accG[MF][NF] = {};
  for (int k0 = 0; k0 < HDIM; k0 += 32) {
    v16h au[MF], ah[MF];
#pragma unroll
    for (int mf = 0; mf < MF; ++mf) {
      au[mf] = load_frag16(u, row0 + mf * 16, k0, HDIM, lane);
      ah[mf] = load_frag16(h, row0 + mf * 16, k0, HDIM, lane);
    }
#pragma unroll
    for (int nf = 0; nf < NF; ++nf) {
      v16h bf2 = load_frag16(W2, n0 + nf * 16, k0, HDIM, lane);
      v16h bfg = load_frag16(Wg, n0 + nf * 16, k0, HDIM, lane);
#pragma unroll
      for (int mf = 0; mf < MF; ++mf) {
        accF[mf][nf] = __builtin_amdgcn_wmma_f32_16x16x32_f16(
            false, au[mf], false, bf2, (short)0, accF[mf][nf], false, false);
        accG[mf][nf] = __builtin_amdgcn_wmma_f32_16x16x32_f16(
            false, ah[mf], false, bfg, (short)0, accG[mf][nf], false, false);
      }
    }
  }
#pragma unroll
  for (int mf = 0; mf < MF; ++mf) {
#pragma unroll
    for (int nf = 0; nf < NF; ++nf) {
      const int n = n0 + nf * 16 + (lane & 15);
      const float bb2 = b2[n], bbg = bg[n];
#pragma unroll
      for (int e = 0; e < 8; ++e) {
        const int m = row0 + mf * 16 + e + ((lane >> 4) << 3);
        const int s = m >> 7;
        const int b = m & 127;
        const float dw = noise[(((size_t)s * 23 + t) * 128 + b) * HDIM + n] * sqdt;
        const float f  = accF[mf][nf][e] + bb2;
        const float g  = softplusf_(accG[mf][nf][e] + bbg);
        const float ho = (float)h[(size_t)m * HDIM + n];
        hout[(size_t)m * HDIM + n] = (_Float16)(ho + f * dtv + g * dw);
      }
    }
  }
}

// ---------------------------------------------------------------------------
// Output projection for decoder step t:
//   out[(m*32+n)*24 + t] = (h @ Wout^T)[m][n] + out_b[n]
// ---------------------------------------------------------------------------
__global__ __launch_bounds__(256)
void k_outproj(const _Float16* __restrict__ h, const _Float16* __restrict__ Wout,
               const float* __restrict__ bout, float* __restrict__ out, int t) {
  const int lane = threadIdx.x & 31;
  const int w    = threadIdx.x >> 5;
  const int row0 = (blockIdx.y * 8 + w) * 16;

  v8f acc[2] = {};
  for (int k0 = 0; k0 < HDIM; k0 += 32) {
    v16h a = load_frag16(h, row0, k0, HDIM, lane);
#pragma unroll
    for (int nf = 0; nf < 2; ++nf) {
      v16h bf = load_frag16(Wout, nf * 16, k0, HDIM, lane);
      acc[nf] = __builtin_amdgcn_wmma_f32_16x16x32_f16(
          false, a, false, bf, (short)0, acc[nf], false, false);
    }
  }
#pragma unroll
  for (int nf = 0; nf < 2; ++nf) {
    const int n = nf * 16 + (lane & 15);
    const float bb = bout[n];
#pragma unroll
    for (int e = 0; e < 8; ++e) {
      const int m = row0 + e + ((lane >> 4) << 3);
      out[((size_t)m * 32 + n) * 24 + t] = acc[nf][e] + bb;
    }
  }
}

// ---------------------------------------------------------------------------
// Host-side orchestration
// ---------------------------------------------------------------------------
extern "C" void kernel_launch(void* const* d_in, const int* in_sizes, int n_in,
                              void* d_out, int out_size, void* d_ws, size_t ws_size,
                              hipStream_t stream) {
  (void)in_sizes; (void)n_in; (void)out_size; (void)ws_size;

  const float* x       = (const float*)d_in[0];
  const float* noise   = (const float*)d_in[1];
  const float* enc_Wih = (const float*)d_in[2];
  const float* enc_Whh = (const float*)d_in[3];
  const float* enc_bih = (const float*)d_in[4];
  const float* enc_bhh = (const float*)d_in[5];
  const float* f_W1    = (const float*)d_in[6];
  const float* f_b1    = (const float*)d_in[7];
  const float* f_W2    = (const float*)d_in[8];
  const float* f_b2    = (const float*)d_in[9];
  const float* g_W     = (const float*)d_in[10];
  const float* g_b     = (const float*)d_in[11];
  /* d_in[12] = dec_Wih: unused (decoder inputs are zero) */
  const float* dec_Whh = (const float*)d_in[13];
  const float* dec_bih = (const float*)d_in[14];
  const float* dec_bhh = (const float*)d_in[15];
  const float* out_W   = (const float*)d_in[16];
  const float* out_b   = (const float*)d_in[17];
  float* out = (float*)d_out;

  _Float16* base = (_Float16*)d_ws;
  size_t off = 0;
  auto take = [&](size_t n) -> _Float16* {
    _Float16* p = base + off;
    off += (n + 127) & ~(size_t)127;
    return p;
  };
  _Float16* wWihE = take((size_t)1536 * 32);
  _Float16* wWhhE = take((size_t)1536 * 512);
  _Float16* wF1   = take((size_t)512 * 512);
  _Float16* wF2   = take((size_t)512 * 512);
  _Float16* wG    = take((size_t)512 * 512);
  _Float16* wWhhD = take((size_t)1536 * 512);
  _Float16* wOut  = take((size_t)32 * 512);
  _Float16* xT    = take((size_t)96 * 128 * 32);
  _Float16* hA    = take((size_t)2048 * 512);
  _Float16* hB    = take((size_t)2048 * 512);
  _Float16* uB    = take((size_t)2048 * 512);

  auto cvt = [&](const float* s, _Float16* d, int n) {
    k_f32_to_f16<<<dim3((n + 255) / 256), 256, 0, stream>>>(s, d, n);
  };
  cvt(enc_Wih, wWihE, 1536 * 32);
  cvt(enc_Whh, wWhhE, 1536 * 512);
  cvt(f_W1, wF1, 512 * 512);
  cvt(f_W2, wF2, 512 * 512);
  cvt(g_W, wG, 512 * 512);
  cvt(dec_Whh, wWhhD, 1536 * 512);
  cvt(out_W, wOut, 32 * 512);
  k_x_transpose<<<dim3((96 * 128 * 32 + 255) / 256), 256, 0, stream>>>(x, xT);
  k_zero_f16<<<dim3((128 * 512) / 256), 256, 0, stream>>>(hA, 128 * 512);

  // ---- Encoder GRU: 96 steps over M=128 (WG tile 128x32, grid 16x1) ----
  _Float16* hc = hA;
  _Float16* hn = hB;
  for (int t = 0; t < 96; ++t) {
    k_gru_step<true, 1, 2, 8, 1><<<dim3(16, 1), 256, 0, stream>>>(
        hc, hn, wWhhE, enc_bih, enc_bhh, xT + (size_t)t * 128 * 32, wWihE);
    _Float16* tmp = hc; hc = hn; hn = tmp;
  }
  // Broadcast h_enc (128 rows) to S*B = 2048 rows.
  k_broadcast<<<dim3((2048 * 512) / 256), 256, 0, stream>>>(hc, hn);
  { _Float16* tmp = hc; hc = hn; hn = tmp; }

  // ---- Euler-Maruyama SDE: 23 steps over M=2048 (grid 8x16) ----
  const float dtv  = 24.0f / 23.0f;
  const float sqdt = sqrtf(dtv);
  for (int t = 0; t < 23; ++t) {
    k_sde_u<2, 2, 4, 2><<<dim3(8, 16), 256, 0, stream>>>(hc, uB, wF1, f_b1);
    k_sde_h<2, 2, 4, 2><<<dim3(8, 16), 256, 0, stream>>>(
        hc, uB, hn, wF2, f_b2, wG, g_b, noise, t, dtv, sqdt);
    _Float16* tmp = hc; hc = hn; hn = tmp;
  }

  // ---- Decoder GRU (TDM-staged) + output projection: 24 steps ----
  for (int t = 0; t < 24; ++t) {
    k_gru_step_tdm<<<dim3(8, 16), 256, 0, stream>>>(hc, hn, wWhhD, dec_bih, dec_bhh);
    k_outproj<<<dim3(1, 16), 256, 0, stream>>>(hn, wOut, out_b, out, t);
    _Float16* tmp = hc; hc = hn; hn = tmp;
  }
}
